// PilotNet_13116830122612
// MI455X (gfx1250) — 8-bit matrix operand order repaired and checked by reading, compiled, not run on hardware
//
#include <hip/hip_runtime.h>
#include <hip/hip_bf16.h>
#include <cstdint>

typedef __attribute__((ext_vector_type(8))) int v8i;

#define BATCH 256

struct Scal {
  unsigned int maxabs[10];
  int maxm[9];
  float sw[9];
  float s[9];
};


__global__ void init_kernel(Scal* sc) {
  const int t = threadIdx.x;
  if (t < 10) sc->maxabs[t] = 0u;
  if (t < 9)  sc->maxm[t] = 0;
}

__global__ void absmax_kernel(const float* __restrict__ x, long long n,
                              unsigned int* __restrict__ dst) {
  __shared__ unsigned int sm;
  if (threadIdx.x == 0) sm = 0u;
  __syncthreads();
  unsigned int m = 0u;
  for (long long i = (long long)blockIdx.x * blockDim.x + threadIdx.x; i < n;
       i += (long long)gridDim.x * blockDim.x) {
    const unsigned int b = __float_as_uint(fabsf(x[i]));
    m = (b > m) ? b : m;
  }
  atomicMax(&sm, m);
  __syncthreads();
  if (threadIdx.x == 0) atomicMax(dst, sm);
}

__global__ void absmax4_kernel(const float4* __restrict__ x, long long n4,
                               unsigned int* __restrict__ dst) {
  __shared__ unsigned int sm;
  if (threadIdx.x == 0) sm = 0u;
  __syncthreads();
  unsigned int m = 0u;
  for (long long i = (long long)blockIdx.x * blockDim.x + threadIdx.x; i < n4;
       i += (long long)gridDim.x * blockDim.x) {
    const float4 v = x[i];
    unsigned int b;
    b = __float_as_uint(fabsf(v.x)); m = (b > m) ? b : m;
    b = __float_as_uint(fabsf(v.y)); m = (b > m) ? b : m;
    b = __float_as_uint(fabsf(v.z)); m = (b > m) ? b : m;
    b = __float_as_uint(fabsf(v.w)); m = (b > m) ? b : m;
  }
  atomicMax(&sm, m);
  __syncthreads();
  if (threadIdx.x == 0) atomicMax(dst, sm);
}

__global__ void wquant_perm_kernel(const float* __restrict__ w, int oc_n, int ktot,
                                   int ic_n, int khw, int kpitch, long long n,
                                   const unsigned int* __restrict__ maxabs,
                                   float* __restrict__ sw_out,
                                   signed char* __restrict__ wq) {
  const float s = fmaxf(__uint_as_float(*maxabs) / 7.0f, 1e-8f);
  if (blockIdx.x == 0 && threadIdx.x == 0) *sw_out = s;
  const float inv = 1.0f / s;
  for (long long i = (long long)blockIdx.x * blockDim.x + threadIdx.x; i < n;
       i += (long long)gridDim.x * blockDim.x) {
    const int oc = (int)(i / kpitch);
    const int k  = (int)(i - (long long)oc * kpitch);
    signed char r = 0;
    if (oc < oc_n && k < ktot) {
      const int cell = k / ic_n;
      const int ic   = k - cell * ic_n;
      const long long src = (long long)oc * ktot + (long long)ic * khw + cell;
      const float q = fminf(fmaxf(rintf(w[src] * inv), -7.0f), 7.0f);
      r = (signed char)(int)q;
    }
    wq[i] = r;
  }
}

__device__ __forceinline__ unsigned int pack_i8(float q) {
  return (unsigned int)(unsigned char)(signed char)(int)q;
}

__global__ void xquant_kernel(const float4* __restrict__ x, long long n4,
                              const unsigned int* __restrict__ maxabs,
                              float* __restrict__ s_out,
                              unsigned int* __restrict__ aq) {
  const float s = fmaxf(__uint_as_float(*maxabs) * (1.0f / 255.0f) * (1.0f / 127.0f), 1e-8f);
  if (blockIdx.x == 0 && threadIdx.x == 0) *s_out = s;
  const float inv = (1.0f / 255.0f) / s;
  for (long long i = (long long)blockIdx.x * blockDim.x + threadIdx.x; i < n4;
       i += (long long)gridDim.x * blockDim.x) {
    const float4 v = x[i];
    const float qx = fminf(fmaxf(rintf(v.x * inv), -128.0f), 127.0f);
    const float qy = fminf(fmaxf(rintf(v.y * inv), -128.0f), 127.0f);
    const float qz = fminf(fmaxf(rintf(v.z * inv), -128.0f), 127.0f);
    const float qw = fminf(fmaxf(rintf(v.w * inv), -128.0f), 127.0f);
    aq[i] = pack_i8(qx) | (pack_i8(qy) << 8) | (pack_i8(qz) << 16) | (pack_i8(qw) << 24);
  }
}

__global__ void bquant_kernel(const float* __restrict__ b, int n,
                              const float* __restrict__ s_prev,
                              const float* __restrict__ sw,
                              int* __restrict__ bi) {
  const float inv = 1.0f / ((*s_prev) * (*sw));
  const int i = blockIdx.x * blockDim.x + threadIdx.x;
  if (i < n) {
    const float q = fminf(fmaxf(rintf(b[i] * inv), -128.0f), 127.0f);
    bi[i] = (int)q;
  }
}

__global__ void biasfill_kernel(int* __restrict__ m, long long n, int ocn,
                                const int* __restrict__ bi) {
  for (long long i = (long long)blockIdx.x * blockDim.x + threadIdx.x; i < n;
       i += (long long)gridDim.x * blockDim.x) {
    m[i] = bi[(int)(i % ocn)];
  }
}

__global__ void relumax_kernel(const int* __restrict__ m, long long n,
                               int* __restrict__ dst) {
  __shared__ int sm;
  if (threadIdx.x == 0) sm = 0;
  __syncthreads();
  int loc = 0;
  for (long long i = (long long)blockIdx.x * blockDim.x + threadIdx.x; i < n;
       i += (long long)gridDim.x * blockDim.x) {
    const int v = m[i];
    loc = (v > loc) ? v : loc;
  }
  atomicMax(&sm, loc);
  __syncthreads();
  if (threadIdx.x == 0) atomicMax(dst, sm);
}

__global__ void supdate_kernel(const int* __restrict__ maxm,
                               const float* __restrict__ s_prev,
                               const float* __restrict__ sw,
                               float* __restrict__ s_out) {
  int mm = *maxm; if (mm < 0) mm = 0;
  const float maxy = (float)mm * (*s_prev) * (*sw);
  *s_out = fmaxf(maxy * (1.0f / 15.0f), 1e-8f);
}

__device__ __forceinline__ unsigned int q4(int v, float inv, int mm) {
  if (v <= 0 || mm <= 0) return 0u;
  int q = (int)rintf((float)v * inv);
  return (unsigned int)(q > 15 ? 15 : q);
}

__global__ void requant_kernel(const int4* __restrict__ m, long long n4,
                               const int* __restrict__ maxm,
                               unsigned int* __restrict__ out) {
  const int mm = *maxm;
  const float inv = (mm > 0) ? 15.0f / (float)mm : 0.0f;
  for (long long i = (long long)blockIdx.x * blockDim.x + threadIdx.x; i < n4;
       i += (long long)gridDim.x * blockDim.x) {
    const int4 v = m[i];
    out[i] = q4(v.x, inv, mm) | (q4(v.y, inv, mm) << 8) |
             (q4(v.z, inv, mm) << 16) | (q4(v.w, inv, mm) << 24);
  }
}

__global__ void finalout_kernel(const int* __restrict__ m, int n,
                                const float* __restrict__ s_prev,
                                const float* __restrict__ sw,
                                float* __restrict__ out) {
  const float sc = (*s_prev) * (*sw);
  const int i = blockIdx.x * blockDim.x + threadIdx.x;
  if (i < n) {
    const float y = (float)m[i] * sc;
    const float q = fminf(fmaxf(rintf(y * 16384.0f), -16384.0f), 16384.0f);
    out[i] = q * (1.0f / 16384.0f);
  }
}


template <int IC, int IH, int IW, int OC, int OH, int OW, int KH, int KW,
          int ST, bool SA, bool SPLITK>
__global__ __launch_bounds__(256) void qconv_wmma(
    const signed char* __restrict__ act, const signed char* __restrict__ wq,
    const int* __restrict__ bias, int* __restrict__ mout,
    int* __restrict__ gmax, int kcpb) {
  constexpr int KTOT   = IC * KH * KW;
  constexpr int KC     = (KTOT + 63) / 64;
  constexpr int KPITCH = KC * 64;
  constexpr int OHW    = OH * OW;
  constexpr int NT     = (OC + 15) / 16;
  constexpr int MT     = 2;
  constexpr bool VEC8  = (IC % 8) == 0;
  constexpr bool VEC4  = (IC % 4) == 0;

  __shared__ int blkmax;
  if (!SPLITK) {
    if (threadIdx.x == 0) blkmax = 0;
    __syncthreads();
  }

  const int wave = threadIdx.x >> 5;
  const int lane = threadIdx.x & 31;
  const int row  = lane & 15;
  const int half = lane >> 4;

  const int ptile0 = (blockIdx.x * 8 + wave) * MT;

  const signed char* pix[MT];
#pragma unroll
  for (int mt = 0; mt < MT; ++mt) {
    const int p    = (ptile0 + mt) * 16 + row;
    const int bidx = p / OHW;
    const int prem = p - bidx * OHW;
    const int oh0  = (prem / OW) * ST;
    const int ow0  = (prem - (prem / OW) * OW) * ST;
    pix[mt] = act + (((long long)bidx * IH + oh0) * IW + ow0) * IC;
  }

  const signed char* bbase[NT];
#pragma unroll
  for (int nt = 0; nt < NT; ++nt)
    bbase[nt] = wq + (long long)(nt * 16 + row) * KPITCH + half * 16;

  const int kc0 = blockIdx.z * kcpb;
  int kc1 = kc0 + kcpb;
  if (kc1 > KC) kc1 = KC;

  const v8i vzero = {};
  v8i acc[MT][NT];
#pragma unroll
  for (int mt = 0; mt < MT; ++mt)
#pragma unroll
    for (int nt = 0; nt < NT; ++nt) acc[mt][nt] = vzero;

  for (int kc = kc0; kc < kc1; ++kc) {
    const int kb = kc * 64;

    v8i af[MT];
    if (VEC8) {
#pragma unroll
      for (int t = 0; t < 4; ++t) {
        const int kj   = kb + half * 16 + ((t >> 1) << 5) + ((t & 1) << 3);
        const int cell = kj / IC;
        const int icb  = kj - cell * IC;
        const int kh   = cell / KW;
        const int kw   = cell - kh * KW;
        const int off  = (kh * IW + kw) * IC + icb;
#pragma unroll
        for (int mt = 0; mt < MT; ++mt) {
          const int2 d = *(const int2*)(pix[mt] + off);
          af[mt][2 * t]     = d.x;
          af[mt][2 * t + 1] = d.y;
        }
      }
    } else if (VEC4) {
#pragma unroll
      for (int j = 0; j < 8; ++j) {
        const int kj   = kb + half * 16 + ((j >> 2) << 5) + ((j & 3) << 2);
        const int cell = kj / IC;
        const int icb  = kj - cell * IC;
        const int kh   = cell / KW;
        const int kw   = cell - kh * KW;
        const int off  = (kh * IW + kw) * IC + icb;
#pragma unroll
        for (int mt = 0; mt < MT; ++mt)
          af[mt][j] = *(const int*)(pix[mt] + off);
      }
    } else {
#pragma unroll
      for (int j = 0; j < 8; ++j) {
        const int kj = kb + half * 16 + ((j >> 2) << 5) + ((j & 3) << 2);
        unsigned int v0 = 0, v1 = 0;
#pragma unroll
        for (int bb = 0; bb < 4; ++bb) {
          const int k    = kj + bb;
          const int cell = k / IC;
          const int ic   = k - cell * IC;
          const int kh   = cell / KW;
          const int kw   = cell - kh * KW;
          const int off  = (kh * IW + kw) * IC + ic;
          v0 |= ((unsigned int)(unsigned char)pix[0][off]) << (8 * bb);
          v1 |= ((unsigned int)(unsigned char)pix[1][off]) << (8 * bb);
        }
        af[0][j] = (int)v0;
        af[1][j] = (int)v1;
      }
    }

#pragma unroll
    for (int nt = 0; nt < NT; ++nt) {
      const int4 lo = *(const int4*)(bbase[nt] + kb);
      const int4 hi = *(const int4*)(bbase[nt] + kb + 32);
      v8i bf;
      bf[0] = lo.x; bf[1] = lo.y; bf[2] = lo.z; bf[3] = lo.w;
      bf[4] = hi.x; bf[5] = hi.y; bf[6] = hi.z; bf[7] = hi.w;
#pragma unroll
      for (int mt = 0; mt < MT; ++mt)
        acc[mt][nt] = __builtin_amdgcn_wmma_i32_16x16x64_iu8(
            SA, af[mt], true, bf, acc[mt][nt], false, false);
    }
  }

  int locmax = 0;
#pragma unroll
  for (int mt = 0; mt < MT; ++mt) {
#pragma unroll
    for (int nt = 0; nt < NT; ++nt) {
      const int oc = nt * 16 + row;
      if (oc < OC) {
        const int bv = SPLITK ? 0 : bias[oc];
#pragma unroll
        for (int j = 0; j < 8; ++j) {
          const int prow = (ptile0 + mt) * 16 + j + half * 8;
          const long long idx = (long long)prow * OC + oc;
          if (SPLITK) {
            atomicAdd(&mout[idx], acc[mt][nt][j]);
          } else {
            const int val = acc[mt][nt][j] + bv;
            mout[idx] = val;
            locmax = (val > locmax) ? val : locmax;
          }
        }
      }
    }
  }
  if (!SPLITK) {
    atomicMax(&blkmax, locmax);
    __syncthreads();
    if (threadIdx.x == 0) atomicMax(gmax, blkmax);
  }
}

template <int IC, int IH, int IW, int OC, int OH, int OW, int KH, int KW,
          int ST, bool SA, bool SPLITK>
static void run_conv(const signed char* act, const signed char* w, const int* bias,
                     int* mout, int* gmax, int zsplit, hipStream_t stream) {
  constexpr int OHW  = OH * OW;
  constexpr int KTOT = IC * KH * KW;
  constexpr int KC   = (KTOT + 63) / 64;
  const int ntile = (BATCH * OHW) / 16;
  dim3 grid(ntile / 16, 1, SPLITK ? zsplit : 1);
  const int kcpb = SPLITK ? (KC + zsplit - 1) / zsplit : KC;
  qconv_wmma<IC, IH, IW, OC, OH, OW, KH, KW, ST, SA, SPLITK>
      <<<grid, 256, 0, stream>>>(act, w, bias, mout, gmax, kcpb);
}


extern "C" void kernel_launch(void* const* d_in, const int* in_sizes, int n_in,
                              void* d_out, int out_size, void* d_ws, size_t ws_size,
                              hipStream_t stream) {
  (void)in_sizes; (void)n_in; (void)out_size; (void)ws_size;

  const float* x = (const float*)d_in[0];
  const float* wsrc[9] = {(const float*)d_in[1],  (const float*)d_in[3],
                          (const float*)d_in[5],  (const float*)d_in[7],
                          (const float*)d_in[9],  (const float*)d_in[11],
                          (const float*)d_in[13], (const float*)d_in[15],
                          (const float*)d_in[17]};
  const float* bsrc[9] = {(const float*)d_in[2],  (const float*)d_in[4],
                          (const float*)d_in[6],  (const float*)d_in[8],
                          (const float*)d_in[10], (const float*)d_in[12],
                          (const float*)d_in[14], (const float*)d_in[16],
                          (const float*)d_in[18]};
  float* out = (float*)d_out;

  char* ws = (char*)d_ws;
  Scal* sc = (Scal*)ws;
  size_t cur = 256;
  auto alloc = [&](size_t bytes) -> char* {
    char* p = ws + cur;
    cur += (bytes + 255) & ~(size_t)255;
    return p;
  };

  static const int ocn[9]     = {24, 36, 48, 64, 64, 100, 50, 10, 1};
  static const int ktot_t[9]  = {25, 600, 900, 432, 576, 48576, 100, 50, 10};
  static const int icn_t[9]   = {1, 24, 36, 48, 64, 64, 100, 50, 10};
  static const int khw_t[9]   = {25, 25, 25, 9, 9, 759, 1, 1, 1};
  static const int kpitch_t[9] = {64, 640, 960, 448, 576, 48576, 128, 64, 64};
  static const int ocpad_t[9]  = {32, 48, 48, 64, 64, 112, 64, 16, 16};
  static const long long wn[9] = {600, 21600, 43200, 27648, 36864, 4857600, 5000, 500, 10};

  signed char* wq[9];
  int* bi[9];
  for (int i = 0; i < 9; ++i)
    wq[i] = (signed char*)alloc((size_t)ocpad_t[i] * kpitch_t[i]);
  for (int i = 0; i < 9; ++i) bi[i] = (int*)alloc((size_t)ocn[i] * 4);

  const size_t ACT_MAX = 114548736ull;
  signed char* actA = (signed char*)alloc(ACT_MAX);
  signed char* actB = (signed char*)alloc(ACT_MAX);
  int* mbuf = (int*)alloc(ACT_MAX * 4);

  init_kernel<<<1, 32, 0, stream>>>(sc);
  for (int i = 0; i < 9; ++i) {
    int blocks = (int)((wn[i] + 255) / 256);
    if (blocks > 2048) blocks = 2048;
    absmax_kernel<<<blocks, 256, 0, stream>>>(wsrc[i], wn[i], &sc->maxabs[1 + i]);
  }
  const long long NX4 = 256ll * 240 * 320 / 4;
  absmax4_kernel<<<2048, 256, 0, stream>>>((const float4*)x, NX4, &sc->maxabs[0]);
  for (int i = 0; i < 9; ++i) {
    const long long n = (long long)ocpad_t[i] * kpitch_t[i];
    int blocks = (int)((n + 255) / 256);
    if (blocks > 4096) blocks = 4096;
    wquant_perm_kernel<<<blocks, 256, 0, stream>>>(wsrc[i], ocn[i], ktot_t[i],
                                                   icn_t[i], khw_t[i], kpitch_t[i], n,
                                                   &sc->maxabs[1 + i], &sc->sw[i], wq[i]);
  }
  xquant_kernel<<<2048, 256, 0, stream>>>((const float4*)x, NX4, &sc->maxabs[0],
                                          &sc->s[0], (unsigned int*)actA);

  static const long long nact[8] = {114548736ll, 40449024ll, 12275712ll, 14336000ll,
                                    12435456ll,  25600ll,    12800ll,    2560ll};

  bquant_kernel<<<1, 128, 0, stream>>>(bsrc[0], ocn[0], &sc->s[0], &sc->sw[0], bi[0]);
  run_conv<1, 240, 320, 24, 118, 158, 5, 5, 2, true, false>(actA, wq[0], bi[0], mbuf,
                                                            &sc->maxm[0], 1, stream);
  supdate_kernel<<<1, 1, 0, stream>>>(&sc->maxm[0], &sc->s[0], &sc->sw[0], &sc->s[1]);
  requant_kernel<<<2048, 256, 0, stream>>>((const int4*)mbuf, nact[0] / 4, &sc->maxm[0],
                                           (unsigned int*)actB);

  bquant_kernel<<<1, 128, 0, stream>>>(bsrc[1], ocn[1], &sc->s[1], &sc->sw[1], bi[1]);
  run_conv<24, 118, 158, 36, 57, 77, 5, 5, 2, false, false>(actB, wq[1], bi[1], mbuf,
                                                            &sc->maxm[1], 1, stream);
  supdate_kernel<<<1, 1, 0, stream>>>(&sc->maxm[1], &sc->s[1], &sc->sw[1], &sc->s[2]);
  requant_kernel<<<2048, 256, 0, stream>>>((const int4*)mbuf, nact[1] / 4, &sc->maxm[1],
                                           (unsigned int*)actA);

  bquant_kernel<<<1, 128, 0, stream>>>(bsrc[2], ocn[2], &sc->s[2], &sc->sw[2], bi[2]);
  run_conv<36, 57, 77, 48, 27, 37, 5, 5, 2, false, false>(actA, wq[2], bi[2], mbuf,
                                                          &sc->maxm[2], 1, stream);
  supdate_kernel<<<1, 1, 0, stream>>>(&sc->maxm[2], &sc->s[2], &sc->sw[2], &sc->s[3]);
  requant_kernel<<<2048, 256, 0, stream>>>((const int4*)mbuf, nact[2] / 4, &sc->maxm[2],
                                           (unsigned int*)actB);

  bquant_kernel<<<1, 128, 0, stream>>>(bsrc[3], ocn[3], &sc->s[3], &sc->sw[3], bi[3]);
  run_conv<48, 27, 37, 64, 25, 35, 3, 3, 1, false, false>(actB, wq[3], bi[3], mbuf,
                                                          &sc->maxm[3], 1, stream);
  supdate_kernel<<<1, 1, 0, stream>>>(&sc->maxm[3], &sc->s[3], &sc->sw[3], &sc->s[4]);
  requant_kernel<<<2048, 256, 0, stream>>>((const int4*)mbuf, nact[3] / 4, &sc->maxm[3],
                                           (unsigned int*)actA);

  bquant_kernel<<<1, 128, 0, stream>>>(bsrc[4], ocn[4], &sc->s[4], &sc->sw[4], bi[4]);
  run_conv<64, 25, 35, 64, 23, 33, 3, 3, 1, false, false>(actA, wq[4], bi[4], mbuf,
                                                          &sc->maxm[4], 1, stream);
  supdate_kernel<<<1, 1, 0, stream>>>(&sc->maxm[4], &sc->s[4], &sc->sw[4], &sc->s[5]);
  requant_kernel<<<2048, 256, 0, stream>>>((const int4*)mbuf, nact[4] / 4, &sc->maxm[4],
                                           (unsigned int*)actB);

  bquant_kernel<<<1, 128, 0, stream>>>(bsrc[5], ocn[5], &sc->s[5], &sc->sw[5], bi[5]);
  biasfill_kernel<<<128, 256, 0, stream>>>(mbuf, nact[5], ocn[5], bi[5]);
  run_conv<64, 23, 33, 100, 1, 1, 23, 33, 1, false, true>(actB, wq[5], bi[5], mbuf,
                                                          &sc->maxm[5], 16, stream);
  relumax_kernel<<<128, 256, 0, stream>>>(mbuf, nact[5], &sc->maxm[5]);
  supdate_kernel<<<1, 1, 0, stream>>>(&sc->maxm[5], &sc->s[5], &sc->sw[5], &sc->s[6]);
  requant_kernel<<<128, 256, 0, stream>>>((const int4*)mbuf, nact[5] / 4, &sc->maxm[5],
                                          (unsigned int*)actA);

  bquant_kernel<<<1, 128, 0, stream>>>(bsrc[6], ocn[6], &sc->s[6], &sc->sw[6], bi[6]);
  run_conv<100, 1, 1, 50, 1, 1, 1, 1, 1, false, false>(actA, wq[6], bi[6], mbuf,
                                                       &sc->maxm[6], 1, stream);
  supdate_kernel<<<1, 1, 0, stream>>>(&sc->maxm[6], &sc->s[6], &sc->sw[6], &sc->s[7]);
  requant_kernel<<<64, 256, 0, stream>>>((const int4*)mbuf, nact[6] / 4, &sc->maxm[6],
                                         (unsigned int*)actB);

  bquant_kernel<<<1, 128, 0, stream>>>(bsrc[7], ocn[7], &sc->s[7], &sc->sw[7], bi[7]);
  run_conv<50, 1, 1, 10, 1, 1, 1, 1, 1, false, false>(actB, wq[7], bi[7], mbuf,
                                                      &sc->maxm[7], 1, stream);
  supdate_kernel<<<1, 1, 0, stream>>>(&sc->maxm[7], &sc->s[7], &sc->sw[7], &sc->s[8]);
  requant_kernel<<<16, 256, 0, stream>>>((const int4*)mbuf, nact[7] / 4, &sc->maxm[7],
                                         (unsigned int*)actA);

  bquant_kernel<<<1, 128, 0, stream>>>(bsrc[8], ocn[8], &sc->s[8], &sc->sw[8], bi[8]);
  run_conv<10, 1, 1, 1, 1, 1, 1, 1, 1, false, false>(actA, wq[8], bi[8], mbuf,
                                                     &sc->maxm[8], 1, stream);
  finalout_kernel<<<1, 256, 0, stream>>>(mbuf, 256, &sc->s[8], &sc->sw[8], out);
}
